// GDQN_Attention_72851235275288
// MI455X (gfx1250) — compile-verified, matrix operand order checked by reading, not run on hardware
//
#include <hip/hip_runtime.h>

#define BGRAPH 256
#define NN_    1024
#define NNODE  (BGRAPH * NN_)   // 262144
#define INC    32
#define HDIM   128
#define MAXN_  8
#define NEDGE  2097152

typedef __attribute__((ext_vector_type(16))) _Float16 v16h;
typedef __attribute__((ext_vector_type(8)))  float    v8f;
typedef unsigned int u32x4 __attribute__((ext_vector_type(4)));
typedef unsigned int u32x8 __attribute__((ext_vector_type(8)));

__device__ __forceinline__ float selu_f(float x) {
    const float l = 1.0507009873554805f, a = 1.6732632423543772f;
    return x > 0.f ? l * x : l * a * (__expf(x) - 1.f);
}

__device__ __forceinline__ v8f wmma16(v16h a, v16h b, v8f c) {
    // D = A(16x32 f16) x B(32x16 f16) + C(16x16 f32)
    return __builtin_amdgcn_wmma_f32_16x16x32_f16(false, a, false, b, (short)0, c,
                                                  false, false);
}

// Load a 16x32 f16 fragment from LDS/row-major storage.
// Works for A-fragments (row = M-row) and for B-fragments when the B matrix is
// stored transposed (row = N-column). Per ISA layout, lane's halves are two
// contiguous 8-half runs starting at k = kc*32 + 8*(lane>>4) and +16.
// rowstride should be (K + 8) halves: 16B-aligned rows, bank-conflict padded.
__device__ __forceinline__ v16h frag_ld(const _Float16* base, int row, int rowstride,
                                        int kc, int lane) {
    const _Float16* p = base + row * rowstride + kc * 32 + 8 * (lane >> 4);
    v16h r;
    *reinterpret_cast<uint4*>(&r)       = *reinterpret_cast<const uint4*>(p);
    *(reinterpret_cast<uint4*>(&r) + 1) = *reinterpret_cast<const uint4*>(p + 16);
    return r;
}

// B fragment directly from a row-major f32 weight matrix [K x ldn], column n.
__device__ __forceinline__ v16h frag_ld_w32(const float* W, int ldn, int n, int kc,
                                            int lane) {
    int kb = kc * 32 + 8 * (lane >> 4);
    v16h r;
#pragma unroll
    for (int h = 0; h < 16; ++h) {
        int k = kb + (h < 8 ? h : h + 8);
        r[h] = (_Float16)W[k * ldn + n];
    }
    return r;
}

// Tensor Data Mover: DMA a [rows x cols] f16 2-D tile (row-major, stride_f16
// elements between rows) from global into LDS at lds_off, inserting a 16B pad
// after every 256B row (pad_interval code 5 = 64 DWORDs, pad_amount code 3 =
// 4 DWORDs) to match the bank-conflict-padded LDS layout. One issue per wave;
// completion tracked with TENSORcnt.
__device__ __forceinline__ void tdm_load_tile_f16(unsigned lds_off, const void* gptr,
                                                  unsigned rows, unsigned cols,
                                                  unsigned stride_f16) {
    unsigned long long ga = (unsigned long long)gptr;
    u32x4 g0;
    g0[0] = 1u;                                   // count=1 (valid user descriptor)
    g0[1] = lds_off;                              // lds_addr
    g0[2] = (unsigned)ga;                         // global_addr[31:0]
    g0[3] = (unsigned)(ga >> 32) | (2u << 30);    // global_addr[56:32] | type=2
    u32x8 g1;
    g1[0] = (1u << 16)        // data_size = 1 (2 bytes)
          | (1u << 20)        // pad_enable
          | (5u << 22)        // pad_interval: 64 DWORDs (= one 256B row)
          | (3u << 25);       // pad_amount: 4 DWORDs (= 16B)
    g1[1] = (cols & 0xFFFFu) << 16;                           // tensor_dim0 lo16
    g1[2] = ((cols >> 16) & 0xFFFFu) | ((rows & 0xFFFFu) << 16);  // dim0 hi | dim1 lo
    g1[3] = ((rows >> 16) & 0xFFFFu) | ((cols & 0xFFFFu) << 16);  // dim1 hi | tile_dim0
    g1[4] = rows & 0xFFFFu;                                   // tile_dim1 | tile_dim2=0
    g1[5] = stride_f16;                                       // tensor_dim0_stride lo32
    g1[6] = 0u;
    g1[7] = 0u;
    asm volatile("tensor_load_to_lds %0, %1, null, null" :: "s"(g0), "s"(g1) : "memory");
}

// ---------------- degree / normalization ----------------
__global__ void deg_kernel(const long long* __restrict__ ei, float* __restrict__ deg) {
    int e = blockIdx.x * 256 + threadIdx.x;
    if (e < NEDGE) atomicAdd(&deg[(int)ei[(size_t)NEDGE + e]], 1.f);
}

__global__ void dinv_kernel(float* __restrict__ deg) {
    int i = blockIdx.x * 256 + threadIdx.x;
    if (i < NNODE) deg[i] = rsqrtf(deg[i] + 1.f);  // +1 self loop
}

// ---------------- WMMA GEMM: out[M,128] = (f16)(A[M,K] @ B[K,128] + bias) * scale
template <typename AT>
__global__ void gemm_wmma(const AT* __restrict__ A, int K, const float* __restrict__ B,
                          const float* __restrict__ bias, float scale, int M,
                          _Float16* __restrict__ out) {
    __shared__ _Float16 Alds[64 * 136];   // [64 rows][K+8] row-major, padded
    __shared__ _Float16 Blds[128 * 136];  // transposed: [n][K+8], padded
    const int tid = threadIdx.x, lane = tid & 31, wave = tid >> 5;
    const int row0 = blockIdx.x * 64;
    const int ldp = K + 8;

    for (int idx = tid; idx < K * 128; idx += 256) {
        int k = idx >> 7, n = idx & 127;
        Blds[n * ldp + k] = (_Float16)B[idx];
    }
    for (int idx = tid; idx < 64 * K; idx += 256) {
        int r = idx / K, k = idx - r * K;
        int row = row0 + r;
        Alds[r * ldp + k] = (row < M) ? (_Float16)A[(size_t)row * K + k] : (_Float16)0.f;
    }
    __syncthreads();

    const int nn = wave * 16 + (lane & 15);
    v8f acc[4] = {};
    const int kt = K >> 5;
    for (int kc = 0; kc < kt; ++kc) {
        v16h bf = frag_ld(Blds, nn, ldp, kc, lane);
#pragma unroll
        for (int mt = 0; mt < 4; ++mt) {
            v16h af = frag_ld(Alds, mt * 16 + (lane & 15), ldp, kc, lane);
            acc[mt] = wmma16(af, bf, acc[mt]);
        }
    }
    float bn = bias ? bias[nn] : 0.f;
#pragma unroll
    for (int mt = 0; mt < 4; ++mt) {
#pragma unroll
        for (int r = 0; r < 8; ++r) {
            int row = row0 + mt * 16 + r + 8 * (lane >> 4);
            if (row < M) out[(size_t)row * 128 + nn] = (_Float16)((acc[mt][r] + bn) * scale);
        }
    }
}

// ---------------- edge aggregation: one wave per edge, 4 columns per lane
__global__ void edge_kernel(const long long* __restrict__ ei,
                            const float* __restrict__ dinv,
                            const _Float16* __restrict__ xw, float* __restrict__ agg) {
    size_t idx = (size_t)blockIdx.x * 256 + threadIdx.x;
    int e = (int)(idx >> 5);
    int g = (int)(idx & 31);
    int s = (int)ei[e];                     // wave-uniform -> scalar load
    int d = (int)ei[(size_t)NEDGE + e];
    float norm = dinv[s] * dinv[d];
    union { uint2 q; _Float16 h[4]; } cv;
    cv.q = *reinterpret_cast<const uint2*>(xw + (size_t)s * 128 + 4 * g);
    float* dst = agg + (size_t)d * 128 + 4 * g;
#pragma unroll
    for (int i2 = 0; i2 < 4; ++i2) atomicAdd(dst + i2, (float)cv.h[i2] * norm);
}

// h = selu(agg + xw * dinv^2 + bias)
__global__ void node_kernel(const float* __restrict__ agg, const _Float16* __restrict__ xw,
                            const float* __restrict__ dinv, const float* __restrict__ bias,
                            _Float16* __restrict__ h) {
    size_t i = (size_t)blockIdx.x * 256 + threadIdx.x;
    int n = (int)(i >> 7), c = (int)(i & 127);
    float di = dinv[n];
    float v = agg[i] + (float)xw[i] * di * di + bias[c];
    h[i] = (_Float16)selu_f(v);
}

// ---------------- query node selection (stable order by node index, then shuffle)
__global__ void qsel_kernel(const float* __restrict__ x, const long long* __restrict__ shuf,
                            int* __restrict__ qnode) {
    int b = threadIdx.x;  // 256 threads, 1 block: one graph per thread
    int qtmp[8];
#pragma unroll
    for (int m = 0; m < 8; ++m) qtmp[m] = 0;
    int cnt = 0;
    for (int n = 0; n < NN_; ++n) {
        float mv = x[((size_t)b * NN_ + n) * INC + (INC - 3)];
        if (mv > 0.5f && cnt < 8) qtmp[cnt++] = n;
    }
#pragma unroll
    for (int m = 0; m < 8; ++m) {
        int sidx = (int)shuf[b * 8 + m];
        qnode[b * 8 + m] = b * NN_ + qtmp[sidx & 7];
    }
}

__global__ void qgather_kernel(const _Float16* __restrict__ h, const int* __restrict__ qnode,
                               _Float16* __restrict__ q) {
    size_t idx = (size_t)blockIdx.x * 256 + threadIdx.x;  // 2048 rows * 64
    int r = (int)(idx >> 6), t = (int)(idx & 63);
    int node = qnode[r];
    reinterpret_cast<unsigned int*>(q)[(size_t)r * 64 + t] =
        reinterpret_cast<const unsigned int*>(h)[(size_t)node * 64 + t];
}

// ---------------- fused flash-attention + Wo + SELU + final FC, one block/graph
__global__ void attn_kernel(const _Float16* __restrict__ Qb, const _Float16* __restrict__ Kb,
                            const _Float16* __restrict__ Vb, const float* __restrict__ Wo,
                            const float* __restrict__ bo, const float* __restrict__ Wfc,
                            const float* __restrict__ bfc, float* __restrict__ out) {
    __shared__ _Float16 Qs[16 * 136];    // queries (rows 8..15 zero), pre-scaled
    __shared__ _Float16 Ks[16 * 136];    // key tile, [key][feat], TDM-padded rows
    __shared__ _Float16 Vt[128 * 40];    // value tile transposed, [feat][key(pad32)]
    __shared__ _Float16 Ps[16 * 40];     // softmax probs (cols 16..31 zero)
    __shared__ float alphas[16];
    __shared__ float linvs[16];
    __shared__ _Float16 attns[16 * 136];
    __shared__ float selus[8 * 128];
    __shared__ float red[256];

    const int b = blockIdx.x;
    const int tid = threadIdx.x, lane = tid & 31, wave = tid >> 5;
    const int lh = lane >> 4, l15 = lane & 15;

    {   // stage Q vectorized (rows 8..15 zero)
        int r = (tid & 127) >> 4, c0 = (tid & 15) * 8;
        if (tid < 128) {
            *reinterpret_cast<uint4*>(Qs + r * 136 + c0) =
                *reinterpret_cast<const uint4*>(Qb + ((size_t)b * 8 + r) * 128 + c0);
        } else {
            uint4 z = {0u, 0u, 0u, 0u};
            *reinterpret_cast<uint4*>(Qs + (8 + r) * 136 + c0) = z;
        }
    }
    for (int e = tid; e < 128 * 16; e += 256) {  // zero Vt k-pad 16..31
        Vt[(e >> 4) * 40 + 16 + (e & 15)] = (_Float16)0.f;
    }
    for (int e = tid; e < 16 * 16; e += 256) {   // zero Ps cols 16..31
        Ps[(e >> 4) * 40 + 16 + (e & 15)] = (_Float16)0.f;
    }

    float mrun[8], lrun[8];
#pragma unroll
    for (int r = 0; r < 8; ++r) { mrun[r] = -3.0e38f; lrun[r] = 0.f; }
    v8f acc = {};

    for (int ktile = 0; ktile < 64; ++ktile) {
        __syncthreads();
        const size_t base = ((size_t)b * NN_ + ktile * 16) * 128;
        if (wave == 0) {  // TDM: async DMA the 16x128 f16 key tile into padded LDS
            tdm_load_tile_f16((unsigned)(size_t)Ks, Kb + base, 16u, 128u, 128u);
        }
        {   // stage V tile transposed: vector global load + scattered LDS stores
            int r = tid >> 4, c0 = (tid & 15) * 8;
            union { uint4 q; _Float16 hh[8]; } cv;
            cv.q = *reinterpret_cast<const uint4*>(Vb + base + (size_t)r * 128 + c0);
#pragma unroll
            for (int i = 0; i < 8; ++i) Vt[(c0 + i) * 40 + r] = cv.hh[i];
        }
        if (ktile < 63) {  // global_prefetch_b8 next tile
            __builtin_prefetch(Kb + base + 2048 + (tid << 3), 0, 1);
            __builtin_prefetch(Vb + base + 2048 + (tid << 3), 0, 1);
        }
        if (wave == 0) __builtin_amdgcn_s_wait_tensorcnt(0);
        __syncthreads();
        if (wave == 0) {
            v8f s = {};
#pragma unroll
            for (int kc = 0; kc < 4; ++kc) {
                v16h qa  = frag_ld(Qs, l15, 136, kc, lane);
                v16h kbf = frag_ld(Ks, l15, 136, kc, lane);  // B = K^T (keys as rows)
                s = wmma16(qa, kbf, s);
            }
#pragma unroll
            for (int r = 0; r < 8; ++r) {  // online softmax per row within each half
                float v = s[r];
                float mx = v;
#pragma unroll
                for (int d = 1; d < 16; d <<= 1) mx = fmaxf(mx, __shfl_xor(mx, d, 32));
                float mn = fmaxf(mrun[r], mx);
                float al = __expf(mrun[r] - mn);
                float pe = __expf(v - mn);
                float rs = pe;
#pragma unroll
                for (int d = 1; d < 16; d <<= 1) rs += __shfl_xor(rs, d, 32);
                lrun[r] = lrun[r] * al + rs;
                mrun[r] = mn;
                Ps[(r + 8 * lh) * 40 + l15] = (_Float16)pe;
                if (l15 == 0) alphas[r + 8 * lh] = al;
            }
        }
        __syncthreads();
        v16h pa  = frag_ld(Ps, l15, 40, 0, lane);               // A = P (K=32, keys padded)
        v16h vbf = frag_ld(Vt, wave * 16 + l15, 40, 0, lane);   // B = V (transposed store)
#pragma unroll
        for (int r = 0; r < 8; ++r) acc[r] *= alphas[r + 8 * lh];
        acc = wmma16(pa, vbf, acc);
    }

    if (wave == 0 && l15 == 0) {
#pragma unroll
        for (int r = 0; r < 8; ++r) linvs[r + 8 * lh] = 1.f / lrun[r];
    }
    __syncthreads();
    {
        const int nn = wave * 16 + l15;
#pragma unroll
        for (int r = 0; r < 8; ++r) {
            int m = r + 8 * lh;
            attns[m * 136 + nn] = (_Float16)(acc[r] * linvs[m]);
        }
    }
    __syncthreads();
    {   // out = attn @ Wo + bo, then SELU
        const int nn = wave * 16 + l15;
        v8f oc = {};
#pragma unroll
        for (int kc = 0; kc < 4; ++kc) {
            v16h aa = frag_ld(attns, l15, 136, kc, lane);
            v16h wb = frag_ld_w32(Wo, 128, nn, kc, lane);
            oc = wmma16(aa, wb, oc);
        }
        float bn = bo[nn];
#pragma unroll
        for (int r = 0; r < 8; ++r) {
            int m = r + 8 * lh;
            if (m < 8) selus[m * 128 + nn] = selu_f(oc[r] + bn);
        }
    }
    __syncthreads();
    {   // final FC: [1024] @ Wfc[1024,8] + bfc
        int j = tid & 7, part = tid >> 3;
        float sum = 0.f;
        for (int i = part; i < 1024; i += 32) sum += selus[i] * Wfc[i * 8 + j];
        red[tid] = sum;
        __syncthreads();
        for (int off2 = 16; off2 >= 1; off2 >>= 1) {
            if (part < off2) red[j + 8 * part] += red[j + 8 * (part + off2)];
            __syncthreads();
        }
        if (part == 0) out[b * 8 + j] = red[j] + bfc[j];
    }
}

extern "C" void kernel_launch(void* const* d_in, const int* in_sizes, int n_in,
                              void* d_out, int out_size, void* d_ws, size_t ws_size,
                              hipStream_t stream) {
    const float*     x    = (const float*)d_in[0];
    const long long* ei   = (const long long*)d_in[1];
    const long long* shuf = (const long long*)d_in[2];
    const float* W1 = (const float*)d_in[3];
    const float* b1 = (const float*)d_in[4];
    const float* W2 = (const float*)d_in[5];
    const float* b2 = (const float*)d_in[6];
    const float* Wq = (const float*)d_in[7];
    const float* bq = (const float*)d_in[8];
    const float* Wk = (const float*)d_in[9];
    const float* bk = (const float*)d_in[10];
    const float* Wv = (const float*)d_in[11];
    const float* bv = (const float*)d_in[12];
    const float* Wo = (const float*)d_in[13];
    const float* bo = (const float*)d_in[14];
    const float* Wfc = (const float*)d_in[15];
    const float* bfc = (const float*)d_in[16];
    float* out = (float*)d_out;

    char* ws = (char*)d_ws;
    size_t off = 0;
    auto carve = [&](size_t bytes) -> char* {
        char* p = ws + off;
        off += (bytes + 255) & ~(size_t)255;
        return p;
    };
    float*    dinv  = (float*)carve((size_t)NNODE * 4);
    _Float16* xw    = (_Float16*)carve((size_t)NNODE * 128 * 2);
    float*    agg   = (float*)carve((size_t)NNODE * 128 * 4);
    _Float16* h     = (_Float16*)carve((size_t)NNODE * 128 * 2);
    _Float16* kb    = (_Float16*)carve((size_t)NNODE * 128 * 2);
    _Float16* vb    = (_Float16*)carve((size_t)NNODE * 128 * 2);
    int*      qnode = (int*)carve(2048 * 4);
    _Float16* qg    = (_Float16*)carve((size_t)2048 * 128 * 2);
    _Float16* qq    = (_Float16*)carve((size_t)2048 * 128 * 2);

    // degree + normalization
    hipMemsetAsync(dinv, 0, (size_t)NNODE * 4, stream);
    deg_kernel<<<NEDGE / 256, 256, 0, stream>>>(ei, dinv);
    dinv_kernel<<<NNODE / 256, 256, 0, stream>>>(dinv);

    // layer 1
    gemm_wmma<float><<<NNODE / 64, 256, 0, stream>>>(x, INC, W1, nullptr, 1.f, NNODE, xw);
    hipMemsetAsync(agg, 0, (size_t)NNODE * 128 * 4, stream);
    edge_kernel<<<(unsigned)((size_t)NEDGE * 32 / 256), 256, 0, stream>>>(ei, dinv, xw, agg);
    node_kernel<<<(unsigned)((size_t)NNODE * 128 / 256), 256, 0, stream>>>(agg, xw, dinv, b1, h);

    // layer 2
    gemm_wmma<_Float16><<<NNODE / 64, 256, 0, stream>>>(h, HDIM, W2, nullptr, 1.f, NNODE, xw);
    hipMemsetAsync(agg, 0, (size_t)NNODE * 128 * 4, stream);
    edge_kernel<<<(unsigned)((size_t)NEDGE * 32 / 256), 256, 0, stream>>>(ei, dinv, xw, agg);
    node_kernel<<<(unsigned)((size_t)NNODE * 128 / 256), 256, 0, stream>>>(agg, xw, dinv, b2, h);

    // K/V projections over all nodes
    gemm_wmma<_Float16><<<NNODE / 64, 256, 0, stream>>>(h, HDIM, Wk, bk, 1.f, NNODE, kb);
    gemm_wmma<_Float16><<<NNODE / 64, 256, 0, stream>>>(h, HDIM, Wv, bv, 1.f, NNODE, vb);

    // queries: select, gather, project (fold 1/sqrt(H) into Q)
    qsel_kernel<<<1, 256, 0, stream>>>(x, shuf, qnode);
    qgather_kernel<<<2048 * 64 / 256, 256, 0, stream>>>(h, qnode, qg);
    gemm_wmma<_Float16><<<2048 / 64, 256, 0, stream>>>(qg, HDIM, Wq, bq,
                                                       0.08838834764831845f, 2048, qq);

    // fused attention + output head
    attn_kernel<<<BGRAPH, 256, 0, stream>>>(qq, kb, vb, Wo, bo, Wfc, bfc, out);
}
